// SparseLinear_32409823215844
// MI455X (gfx1250) — matrix operand order repaired and checked by reading, not compiled, not run on hardware
//
#include <hip/hip_runtime.h>

// Problem dims (fixed by the reference)
#define N_ROWS 200000
#define KDIM   512
#define ODIM   512

#define ROWS_PER_BLOCK 32            // 2 m-subtiles of 16 per wave
#define STR (KDIM + 8)               // padded LDS row stride (ushorts) -> bank-conflict-free b128 reads

typedef __attribute__((ext_vector_type(16))) __bf16       v16bf;
typedef __attribute__((ext_vector_type(4)))  __bf16       v4bf;
typedef __attribute__((ext_vector_type(8)))  float        v8f;
typedef __attribute__((ext_vector_type(4)))  unsigned int u32x4;

union Frag {
    struct { u32x4 lo, hi; } p;      // two 16-byte runs
    v16bf v;
};

// ---------------------------------------------------------------------------
// Kernel 1: split fp32 weight into bf16 hi/lo planes (bf16x3 decomposition).
// ---------------------------------------------------------------------------
__global__ void SparseLinear_split_weight(const float* __restrict__ w,
                                          __bf16* __restrict__ hi,
                                          __bf16* __restrict__ lo,
                                          int n) {
    int i = blockIdx.x * blockDim.x + threadIdx.x;
    if (i < n) {
        float x  = w[i];
        __bf16 h = (__bf16)x;                 // RNE truncation to bf16
        hi[i] = h;
        lo[i] = (__bf16)(x - (float)h);       // residual
    }
}

// ---------------------------------------------------------------------------
// Kernel 2: GEMM  out[n,o] = sum_k values[n,k] * weight[o,k] + bias[o]
// One block = 32 rows x full 512 cols. 8 waves; wave w owns cols [64w, 64w+64).
// ---------------------------------------------------------------------------
__launch_bounds__(256)
__global__ void SparseLinear_gemm(const float*  __restrict__ values,
                                  const __bf16* __restrict__ wHi,
                                  const __bf16* __restrict__ wLo,
                                  const float*  __restrict__ bias,
                                  float*        __restrict__ out) {
    extern __shared__ unsigned char smem_raw[];
    unsigned short* aHi = (unsigned short*)smem_raw;                 // [ROWS_PER_BLOCK][STR] bf16
    unsigned short* aLo = aHi + ROWS_PER_BLOCK * STR;

    const int tid     = threadIdx.x;
    const int wave    = tid >> 5;
    const int lane    = tid & 31;
    const int r16     = lane & 15;
    const int half    = lane >> 4;
    const int rowBase = blockIdx.x * ROWS_PER_BLOCK;   // 200000 % 32 == 0, no tail

    // ---- Stage A tile: 32x512 fp32 -> bf16 hi/lo in LDS (64 floats/thread) ----
    #pragma unroll
    for (int i = 0; i < (ROWS_PER_BLOCK * KDIM / 4) / 256; ++i) {
        int idx4 = tid + i * 256;
        int e    = idx4 * 4;
        int row  = e >> 9;           // /512
        int k    = e & (KDIM - 1);
        const float4 v = *(const float4*)(values + (size_t)(rowBase + row) * KDIM + k);
        float xs[4] = {v.x, v.y, v.z, v.w};
        v4bf h, l;
        #pragma unroll
        for (int c = 0; c < 4; ++c) {
            __bf16 hb = (__bf16)xs[c];
            h[c] = hb;
            l[c] = (__bf16)(xs[c] - (float)hb);
        }
        *(v4bf*)&aHi[row * STR + k] = h;
        *(v4bf*)&aLo[row * STR + k] = l;
    }
    __syncthreads();

    // ---- Accumulators: 2 m-subtiles x 4 o-subtiles of 16x16 f32 ----
    v8f acc[2][4];
    #pragma unroll
    for (int mt = 0; mt < 2; ++mt)
        #pragma unroll
        for (int ot = 0; ot < 4; ++ot)
            acc[mt][ot] = (v8f){0.f, 0.f, 0.f, 0.f, 0.f, 0.f, 0.f, 0.f};

    // ---- K loop: 16 steps of K=32 ----
    for (int k0 = 0; k0 < KDIM; k0 += 32) {
        // A fragments (16x32 bf16): lane row = r16; contiguous K per lane half (matches B):
        //   half=0: j0..15 -> K=k0..k0+15
        //   half=1: j0..15 -> K=k0+16..k0+31
        v16bf aH[2], aL[2];
        #pragma unroll
        for (int mt = 0; mt < 2; ++mt) {
            int base = (mt * 16 + r16) * STR;
            int o1 = base + k0 + half * 16;
            int o2 = base + k0 + half * 16 + 8;
            Frag f;
            f.p.lo = *(const u32x4*)&aHi[o1];
            f.p.hi = *(const u32x4*)&aHi[o2];
            aH[mt] = f.v;
            f.p.lo = *(const u32x4*)&aLo[o1];
            f.p.hi = *(const u32x4*)&aLo[o2];
            aL[mt] = f.v;
        }
        #pragma unroll
        for (int ot = 0; ot < 4; ++ot) {
            // B fragment (32x16 bf16): lane col = o0 + r16;
            //   lanes 0-15 hold K=k0..k0+15, lanes 16-31 hold K=k0+16..k0+31 (contiguous)
            int col = wave * 64 + ot * 16 + r16;
            int off = col * KDIM + k0 + half * 16;
            v16bf bH = *(const v16bf*)&wHi[off];
            v16bf bL = *(const v16bf*)&wLo[off];
            #pragma unroll
            for (int mt = 0; mt < 2; ++mt) {
                acc[mt][ot] = __builtin_amdgcn_wmma_f32_16x16x32_bf16(
                    false, aH[mt], false, bH, (short)0, acc[mt][ot], false, false);
                acc[mt][ot] = __builtin_amdgcn_wmma_f32_16x16x32_bf16(
                    false, aH[mt], false, bL, (short)0, acc[mt][ot], false, false);
                acc[mt][ot] = __builtin_amdgcn_wmma_f32_16x16x32_bf16(
                    false, aL[mt], false, bH, (short)0, acc[mt][ot], false, false);
            }
        }
    }

    // ---- Epilogue: C/D layout -> row = m0 + half*8 + v, col = o0 + r16 ----
    #pragma unroll
    for (int mt = 0; mt < 2; ++mt) {
        #pragma unroll
        for (int ot = 0; ot < 4; ++ot) {
            int col  = wave * 64 + ot * 16 + r16;
            float bb = bias[col];
            int row0 = rowBase + mt * 16 + half * 8;
            #pragma unroll
            for (int v = 0; v < 8; ++v)
                out[(size_t)(row0 + v) * ODIM + col] = acc[mt][ot][v] + bb;
        }
    }
}

// ---------------------------------------------------------------------------
extern "C" void kernel_launch(void* const* d_in, const int* in_sizes, int n_in,
                              void* d_out, int out_size, void* d_ws, size_t ws_size,
                              hipStream_t stream) {
    (void)in_sizes; (void)n_in; (void)out_size; (void)ws_size;
    const float* values = (const float*)d_in[0];   // [200000, 512] f32
    const float* weight = (const float*)d_in[1];   // [512, 512]    f32
    const float* bias   = (const float*)d_in[2];   // [512]         f32
    // d_in[3] (indices) pass through untouched; reference output is only the GEMM.

    __bf16* wHi = (__bf16*)d_ws;                   // 512 KB
    __bf16* wLo = wHi + (size_t)ODIM * KDIM;       // 512 KB  (ws >= 1 MB)

    const int nW = ODIM * KDIM;
    SparseLinear_split_weight<<<(nW + 255) / 256, 256, 0, stream>>>(weight, wHi, wLo, nW);

    const size_t ldsBytes = (size_t)2 * ROWS_PER_BLOCK * STR * sizeof(unsigned short); // 66,560 B
    SparseLinear_gemm<<<N_ROWS / ROWS_PER_BLOCK, 256, ldsBytes, stream>>>(
        values, wHi, wLo, bias, (float*)d_out);
}
